// torch_GRL_Dueling_2465311228177
// MI455X (gfx1250) — matrix operand order repaired and checked by reading, not compiled, not run on hardware
//
#include <hip/hip_runtime.h>
#include <hip/hip_bf16.h>
#include <stddef.h>

typedef _Float16 v16h __attribute__((ext_vector_type(16)));
typedef float    v8f  __attribute__((ext_vector_type(8)));
typedef float    v4f  __attribute__((ext_vector_type(4)));

#define NN    10000
#define NPAD  10016          // 313 K-tiles * 32
#define HDIM  32
#define FD    128
#define AD    16
#define CHUNK 39             // 312 full K-tiles / 8 waves

static __device__ __forceinline__ _Float16 cv01(float x) {
    return (x != 0.0f) ? (_Float16)1.0f : (_Float16)0.0f;
}

// ---------------------------------------------------------------------------
// Kernel 1: deg_i = count_nonzero(adj[i,:]) + 1 ; dinv_i = rsqrt(deg_i)
// One 256-thread block per row; coalesced v4 NT streaming reads.
// ---------------------------------------------------------------------------
__global__ void deg_kernel(const float* __restrict__ A, float* __restrict__ dinv) {
    __shared__ float sred[256];
    const int row = blockIdx.x;
    const int t   = threadIdx.x;
    const float* ar = A + (size_t)row * NN;
    float s = 0.0f;
    for (int j = t; j < 2500; j += 256) {             // 10000/4 float4 groups
        v4f v = __builtin_nontemporal_load((const v4f*)ar + j);
        s += (v[0] != 0.0f ? 1.0f : 0.0f);
        s += (v[1] != 0.0f ? 1.0f : 0.0f);
        s += (v[2] != 0.0f ? 1.0f : 0.0f);
        s += (v[3] != 0.0f ? 1.0f : 0.0f);
    }
    sred[t] = s;
    __syncthreads();
    for (int o = 128; o > 0; o >>= 1) {
        if (t < o) sred[t] += sred[t + o];
        __syncthreads();
    }
    if (t == 0) dinv[row] = rsqrtf(sred[0] + 1.0f);   // deg >= 1 (self loop)
}

// ---------------------------------------------------------------------------
// Kernel 2: encoder MLP  h = relu(relu(X@W1+b1)@W2+b2)  -> h (f32 ws)
// and Yt[c][row] = (f16)( dinv[row] * (h @ w_g)[row][c] )  (transposed, padded)
// ---------------------------------------------------------------------------
__global__ void enc_kernel(const float* __restrict__ X,
                           const float* __restrict__ w1, const float* __restrict__ b1,
                           const float* __restrict__ w2, const float* __restrict__ b2,
                           const float* __restrict__ wg,
                           const float* __restrict__ dinv,
                           float* __restrict__ hOut, _Float16* __restrict__ yt) {
    __shared__ float sW1[FD * HDIM];
    __shared__ float sW2[HDIM * HDIM];
    __shared__ float sWg[HDIM * HDIM];
    __shared__ float sB1[HDIM], sB2[HDIM];
    __shared__ float sX [8 * FD];
    __shared__ float sH1[8 * HDIM];
    __shared__ float sH2[8 * HDIM];

    const int t = threadIdx.x;
    for (int i = t; i < FD * HDIM; i += 256) sW1[i] = w1[i];
    for (int i = t; i < HDIM * HDIM; i += 256) { sW2[i] = w2[i]; sWg[i] = wg[i]; }
    if (t < HDIM) { sB1[t] = b1[t]; sB2[t] = b2[t]; }

    const int row0 = blockIdx.x * 8;
    for (int i = t; i < 8 * FD; i += 256) {
        int r = i >> 7, k = i & 127;
        int rg = row0 + r;
        sX[i] = (rg < NN) ? X[(size_t)rg * FD + k] : 0.0f;
    }
    __syncthreads();

    const int r  = t >> 5;         // 0..7
    const int c  = t & 31;         // 0..31
    const int rg = row0 + r;

    float s = sB1[c];
    #pragma unroll 4
    for (int k = 0; k < FD; ++k) s += sX[r * FD + k] * sW1[k * HDIM + c];
    sH1[r * HDIM + c] = fmaxf(s, 0.0f);
    __syncthreads();

    s = sB2[c];
    #pragma unroll
    for (int k = 0; k < HDIM; ++k) s += sH1[r * HDIM + k] * sW2[k * HDIM + c];
    s = fmaxf(s, 0.0f);
    sH2[r * HDIM + c] = s;
    if (rg < NN) hOut[(size_t)rg * HDIM + c] = s;
    __syncthreads();

    _Float16 v = (_Float16)0.0f;
    if (rg < NN) {
        float x = 0.0f;
        #pragma unroll
        for (int k = 0; k < HDIM; ++k) x += sH2[r * HDIM + k] * sWg[k * HDIM + c];
        v = (_Float16)(dinv[rg] * x);
    }
    if (rg < NPAD) yt[(size_t)c * NPAD + rg] = v;     // zero tail pad
}

// ---------------------------------------------------------------------------
// Kernel 3: WMMA SpMM + fused dueling head.  625 blocks x 8 waves.
// Wave w owns contiguous K-tiles [39w, 39w+39); wave 7 also takes ragged
// tile 312.  No divergent branches in the hot loop.
// ---------------------------------------------------------------------------
__global__ void gcn_kernel(const float* __restrict__ A,
                           const _Float16* __restrict__ yt,
                           const float* __restrict__ dinv,
                           const float* __restrict__ h,
                           const float* __restrict__ mask,
                           const float* __restrict__ bg,
                           const float* __restrict__ wgd, const float* __restrict__ bgd,
                           const float* __restrict__ wp1, const float* __restrict__ bp1,
                           const float* __restrict__ wp2, const float* __restrict__ bp2,
                           const float* __restrict__ wv,  const float* __restrict__ bv,
                           const float* __restrict__ wa,  const float* __restrict__ ba,
                           float* __restrict__ out) {
    __shared__ float red[8 * 16 * 32];          // per-wave partial tiles (16 KB)
    __shared__ float s_bg[32];
    __shared__ float s_wgd[32 * 32], s_bgd[32];
    __shared__ float s_wp1[64 * 32], s_bp1[32];
    __shared__ float s_wp2[32 * 32], s_bp2[32];
    __shared__ float s_wv[32], s_bv;
    __shared__ float s_wa[32 * 16], s_ba[16];
    __shared__ float s_g1[16 * 32];
    __shared__ float s_p [16 * 64];
    __shared__ float s_adv[16 * 16];
    __shared__ float s_val[16], s_mean[16];

    const int t = threadIdx.x;
    for (int i = t; i < 32 * 32; i += 256) { s_wgd[i] = wgd[i]; s_wp2[i] = wp2[i]; }
    for (int i = t; i < 64 * 32; i += 256) s_wp1[i] = wp1[i];
    for (int i = t; i < 32 * 16; i += 256) s_wa[i] = wa[i];
    if (t < 32) { s_bg[t] = bg[t]; s_bgd[t] = bgd[t]; s_bp1[t] = bp1[t];
                  s_bp2[t] = bp2[t]; s_wv[t] = wv[t]; }
    if (t < 16) s_ba[t] = ba[t];
    if (t == 0) s_bv = bv[0];

    // ---- WMMA main loop -----------------------------------------------
    const int lane = t & 31;
    const int wv8  = t >> 5;            // wave id 0..7
    const int rrow = lane & 15;         // A-tile row this lane feeds
    const int hf   = lane >> 4;         // lane-half (selects K sub-groups)
    const int rowg = blockIdx.x * 16 + rrow;
    const float* Arow = A + (size_t)rowg * NN;

    const int colB = lane & 15;         // B-tile column
    const int kB   = hf * 16;           // B K-base within tile

    // contiguous per-wave streams
    const int k0w = wv8 * CHUNK * 32;   // first K of this wave's chunk
    const float*    aptr = Arow + k0w + hf * 16;
    const _Float16* b0p  = yt + (size_t)colB        * NPAD + kB + k0w;
    const _Float16* b1p  = yt + (size_t)(colB + 16) * NPAD + kB + k0w;

    v8f c0 = {}; v8f c1 = {};
    #pragma unroll 2
    for (int i = 0; i < CHUNK; ++i) {
        __builtin_prefetch(aptr + 512, 0, 0);      // 16 tiles ahead; speculative
        v4f x0 = __builtin_nontemporal_load((const v4f*)(aptr));
        v4f x1 = __builtin_nontemporal_load((const v4f*)(aptr + 4));
        v4f x2 = __builtin_nontemporal_load((const v4f*)(aptr + 8));
        v4f x3 = __builtin_nontemporal_load((const v4f*)(aptr + 12));
        v16h a;
        a[0] =cv01(x0[0]); a[1] =cv01(x0[1]); a[2] =cv01(x0[2]); a[3] =cv01(x0[3]);
        a[4] =cv01(x1[0]); a[5] =cv01(x1[1]); a[6] =cv01(x1[2]); a[7] =cv01(x1[3]);
        a[8] =cv01(x2[0]); a[9] =cv01(x2[1]); a[10]=cv01(x2[2]); a[11]=cv01(x2[3]);
        a[12]=cv01(x3[0]); a[13]=cv01(x3[1]); a[14]=cv01(x3[2]); a[15]=cv01(x3[3]);
        v16h b0 = *(const v16h*)(b0p);
        v16h b1 = *(const v16h*)(b1p);
        c0 = __builtin_amdgcn_wmma_f32_16x16x32_f16(false, a, false, b0, (short)0, c0, false, false);
        c1 = __builtin_amdgcn_wmma_f32_16x16x32_f16(false, a, false, b1, (short)0, c1, false, false);
        aptr += 32; b0p += 32; b1p += 32;
    }

    // ragged K-tile (k0 = 9984, 16 valid columns) -> wave 7 only.
    // Branch is wave-uniform, so EXEC is all-1s inside (WMMA-legal).
    if (wv8 == 7) {
        const int k0 = 312 * 32;
        const int ka = k0 + hf * 16;
        v16h a;
        #pragma unroll
        for (int e = 0; e < 8; ++e) {
            int cA = ka + e, cB2 = ka + 8 + e;
            a[e]     = (cA  < NN) ? cv01(Arow[cA])  : (_Float16)0.0f;
            a[8 + e] = (cB2 < NN) ? cv01(Arow[cB2]) : (_Float16)0.0f;
        }
        v16h b0 = *(const v16h*)(yt + (size_t)colB        * NPAD + kB + k0);
        v16h b1 = *(const v16h*)(yt + (size_t)(colB + 16) * NPAD + kB + k0);
        c0 = __builtin_amdgcn_wmma_f32_16x16x32_f16(false, a, false, b0, (short)0, c0, false, false);
        c1 = __builtin_amdgcn_wmma_f32_16x16x32_f16(false, a, false, b1, (short)0, c1, false, false);
    }

    // spill per-wave partials: C layout -> lanes0-15 hold M=v, lanes16-31 M=v+8
    float* myred = red + wv8 * 512;
    #pragma unroll
    for (int v = 0; v < 8; ++v) {
        int rr = v + 8 * hf;
        myred[rr * 32 + colB]      = c0[v];
        myred[rr * 32 + colB + 16] = c1[v];
    }
    __syncthreads();

    // ---- epilogue: reduce + GCN bias/relu + gd + dueling head ----------
    const int row = t >> 4;            // 0..15
    const int cc  = t & 15;            // 0..15
    const int rg  = blockIdx.x * 16 + row;

    float accA = 0.0f, accB = 0.0f;
    #pragma unroll
    for (int w = 0; w < 8; ++w) {
        accA += red[w * 512 + row * 32 + cc];
        accB += red[w * 512 + row * 32 + cc + 16];
    }
    const float di = dinv[rg];
    // self-loop (+I) term folded in: acc += Yt[c][i] = dinv_i * xw[i][c]
    accA = di * (accA + (float)yt[(size_t)cc        * NPAD + rg]) + s_bg[cc];
    accB = di * (accB + (float)yt[(size_t)(cc + 16) * NPAD + rg]) + s_bg[cc + 16];
    s_g1[row * 32 + cc]      = fmaxf(accA, 0.0f);
    s_g1[row * 32 + cc + 16] = fmaxf(accB, 0.0f);
    __syncthreads();

    // g2 = relu(g1 @ w_gd + b_gd) -> s_p[:,0:32]; h -> s_p[:,32:64]
    #pragma unroll
    for (int j = 0; j < 2; ++j) {
        int c = cc + 16 * j;
        float s = s_bgd[c];
        #pragma unroll
        for (int k = 0; k < 32; ++k) s += s_g1[row * 32 + k] * s_wgd[k * 32 + c];
        s_p[row * 64 + c]      = fmaxf(s, 0.0f);
        s_p[row * 64 + 32 + c] = h[(size_t)rg * 32 + c];
    }
    __syncthreads();

    // p1 = relu(p @ w_p1 + b_p1) -> s_g1 (reuse)
    #pragma unroll
    for (int j = 0; j < 2; ++j) {
        int c = cc + 16 * j;
        float s = s_bp1[c];
        #pragma unroll
        for (int k = 0; k < 64; ++k) s += s_p[row * 64 + k] * s_wp1[k * 32 + c];
        s_g1[row * 32 + c] = fmaxf(s, 0.0f);
    }
    __syncthreads();

    // p2 = relu(p1 @ w_p2 + b_p2) -> s_p[:,0:32] (reuse)
    #pragma unroll
    for (int j = 0; j < 2; ++j) {
        int c = cc + 16 * j;
        float s = s_bp2[c];
        #pragma unroll
        for (int k = 0; k < 32; ++k) s += s_g1[row * 32 + k] * s_wp2[k * 32 + c];
        s_p[row * 64 + c] = fmaxf(s, 0.0f);
    }
    __syncthreads();

    if (cc == 0) {
        float s = s_bv;
        #pragma unroll
        for (int k = 0; k < 32; ++k) s += s_p[row * 64 + k] * s_wv[k];
        s_val[row] = s;
    }
    {
        float s = s_ba[cc];
        #pragma unroll
        for (int k = 0; k < 32; ++k) s += s_p[row * 64 + k] * s_wa[k * 16 + cc];
        s_adv[row * 16 + cc] = s;
    }
    __syncthreads();
    if (cc == 0) {
        float m = 0.0f;
        #pragma unroll
        for (int a2 = 0; a2 < 16; ++a2) m += s_adv[row * 16 + a2];
        s_mean[row] = m * (1.0f / 16.0f);
    }
    __syncthreads();

    const float q = (s_val[row] + s_adv[row * 16 + cc] - s_mean[row]) * mask[rg];
    out[(size_t)rg * 16 + cc] = q;
}

// ---------------------------------------------------------------------------
extern "C" void kernel_launch(void* const* d_in, const int* in_sizes, int n_in,
                              void* d_out, int out_size, void* d_ws, size_t ws_size,
                              hipStream_t stream) {
    (void)in_sizes; (void)n_in; (void)out_size; (void)ws_size;
    const float* feat = (const float*)d_in[0];
    const float* adj  = (const float*)d_in[1];
    const float* mask = (const float*)d_in[2];
    const float* w_e1 = (const float*)d_in[3];
    const float* b_e1 = (const float*)d_in[4];
    const float* w_e2 = (const float*)d_in[5];
    const float* b_e2 = (const float*)d_in[6];
    const float* w_g  = (const float*)d_in[7];
    const float* b_g  = (const float*)d_in[8];
    const float* w_gd = (const float*)d_in[9];
    const float* b_gd = (const float*)d_in[10];
    const float* w_p1 = (const float*)d_in[11];
    const float* b_p1 = (const float*)d_in[12];
    const float* w_p2 = (const float*)d_in[13];
    const float* b_p2 = (const float*)d_in[14];
    const float* w_v  = (const float*)d_in[15];
    const float* b_v  = (const float*)d_in[16];
    const float* w_a  = (const float*)d_in[17];
    const float* b_a  = (const float*)d_in[18];
    float* out = (float*)d_out;

    // workspace layout (64B aligned)
    char* ws = (char*)d_ws;
    float*    dinv = (float*)ws;                                    // NPAD f32
    float*    hbuf = (float*)(ws + 40064);                          // N*32 f32
    _Float16* yt   = (_Float16*)(ws + 40064 + (size_t)NN * 32 * 4); // 32*NPAD f16

    deg_kernel<<<NN, 256, 0, stream>>>(adj, dinv);
    enc_kernel<<<(NPAD + 7) / 8, 256, 0, stream>>>(feat, w_e1, b_e1, w_e2, b_e2,
                                                   w_g, dinv, hbuf, yt);
    gcn_kernel<<<NN / 16, 256, 0, stream>>>(adj, yt, dinv, hbuf, mask, b_g,
                                            w_gd, b_gd, w_p1, b_p1, w_p2, b_p2,
                                            w_v, b_v, w_a, b_a, out);
}
